// architecture_26594437497426
// MI455X (gfx1250) — matrix operand order repaired and checked by reading, not compiled, not run on hardware
//
#include <hip/hip_runtime.h>

// ---------------------------------------------------------------------------
// Fully-fused graph U-Net for MI455X (gfx1250, wave32).
// One workgroup processes BT=8 batch samples end-to-end entirely in LDS.
// SAGE dense layers run on v_wmma_f32_16x16x32_bf16 (f32 accumulate).
// All shape parameters are compile-time templates -> branch-free operand
// builds (cndmask selects, ds_load_b128) and constant-divisor index math.
// ---------------------------------------------------------------------------

#define BT    8
#define NTHR  256
#define NWAVE (NTHR / 32)

typedef __bf16 bf16;
typedef bf16  v16bf __attribute__((ext_vector_type(16)));
typedef float v8f   __attribute__((ext_vector_type(8)));

// LDS layout (floats): bufA | bufB | bufM | aux(320) | warr(960) | wstage(4096)
#define BUFF   20224          // 8 * 316 * 8 floats (largest intermediate)
#define AUXF   320
#define WARRF  960
#define WSTGF  4096           // two bf16 B-layout weight matrices (64x64 max)
#define SMEM_FLOATS (3 * BUFF + AUXF + WARRF + WSTGF)

struct Params {
  const float* x;
  const int*   e[4];   int E[4];     // edge lists (2,E) row-major
  const int*   lab[4];               // pooling labels
  const float* pos[5];               // pos1..pos5, (n,2)
  const int*   ai[4];  int A[4];     // ai_54, ai_43, ai_32, ai_21 (2,A)
  const float* wl[16];               // dims order: 1_1..4_2, 7_1..10_2
  const float* wr[16];
  const float* w5;                   // (8,128)
  const float* w6;                   // (128,8)
  float*       out;                  // (B,256)
};

__device__ __forceinline__ float eluf(float v) { return v > 0.f ? v : __expf(v) - 1.f; }
__device__ __forceinline__ bf16  f2bf(float v) { return (bf16)v; }

__device__ __forceinline__ void zero_lds(float* p, int cnt) {
  for (int i = threadIdx.x; i < cnt; i += NTHR) p[i] = 0.f;
}

// ---------------------------------------------------------------------------
// mean-neighbor aggregation: mbuf[(s*N+dst)*C+c] = mean over in-edges of hin
// ---------------------------------------------------------------------------
template <int N, int C>
static __device__ void aggregate(const float* hin, float* mbuf, float* deg,
                                 const int* edge, int E) {
  constexpr int M = BT * N;
  zero_lds(mbuf, M * C);
  zero_lds(deg, N);
  __syncthreads();
  for (int e = threadIdx.x; e < E; e += NTHR) atomicAdd(&deg[edge[E + e]], 1.f);
  for (int t = threadIdx.x; t < BT * E; t += NTHR) {
    int e = t % E;
    int s = t / E;
    int sn = edge[e], dn = edge[E + e];
    const float* src = &hin[(s * N + sn) * C];
    float*       dst = &mbuf[(s * N + dn) * C];
#pragma unroll
    for (int c = 0; c < C; ++c) atomicAdd(&dst[c], src[c]);
  }
  __syncthreads();
  for (int t = threadIdx.x; t < M; t += NTHR) {
    int node = t % N;                       // constant divisor
    float inv = 1.0f / fmaxf(deg[node], 1.0f);
#pragma unroll
    for (int c = 0; c < C; ++c) mbuf[t * C + c] *= inv;
  }
  __syncthreads();
}

// ---------------------------------------------------------------------------
// hout[M,COUT] = a1[M,CIN] @ w1^T + a2[M,CIN] @ w2^T   (optional ELU)
// Weights are staged into LDS in bf16 B-operand layout, zero-padded to
// KP (mult of 32) x NP (mult of 16), so the hot loop is branch-free.
// ---------------------------------------------------------------------------
template <int N, int CIN, int COUT, bool ELU>
static __device__ void sage_gemm(const float* a1, const float* a2,
                                 const float* w1g, const float* w2g,
                                 float* hout, bf16* wbl, bf16* wbr) {
  constexpr int M  = BT * N;                 // always a multiple of 16 here
  constexpr int NP = (COUT + 15) & ~15;
  constexpr int KP = (CIN + 31) & ~31;
  constexpr int mT = M / 16;
  constexpr int nT = NP / 16;

  // ---- stage weights into LDS, B-layout: wb[k * NP + n] = W[n][k] ----
  for (int t = threadIdx.x; t < KP * NP; t += NTHR) {
    int n = t & (NP - 1);
    int k = t / NP;
    int nc = (n < COUT) ? n : 0;             // clamp -> loads always in bounds
    int kc = (k < CIN)  ? k : 0;
    float vl = w1g[nc * CIN + kc];
    float vr = w2g[nc * CIN + kc];
    bool ok = (n < COUT) && (k < CIN);
    wbl[t] = f2bf(ok ? vl : 0.f);
    wbr[t] = f2bf(ok ? vr : 0.f);
  }
  __syncthreads();

  const int lane = threadIdx.x & 31;
  const int wave = threadIdx.x >> 5;
  const int lrow = lane & 15;                // M-row (A) / N-col (B/D)
  const int lhi  = lane >> 4;

  for (int t = wave; t < mT * nT; t += NWAVE) {
    const int mB = (t / nT) * 16;
    const int nB = (t % nT) * 16;
    v8f acc = {0.f, 0.f, 0.f, 0.f, 0.f, 0.f, 0.f, 0.f};
    const float* pM = a1 + (mB + lrow) * CIN;
    const float* pH = a2 + (mB + lrow) * CIN;
#pragma unroll
    for (int kb = 0; kb < KP; kb += 32) {
      v16bf aM, aH;
      // A 16x32 bf16: lanes 0-15 K={kb..kb+7, kb+16..kb+23}; lanes 16-31 +8.
      // Loads are unconditional (stay inside the LDS allocation); OOB lanes
      // are zeroed with selects (constexpr for CIN multiples of 16/32).
#pragma unroll
      for (int j = 0; j < 8; ++j) {
        const int k0 = kb + lhi * 8 + j;
        const int k1 = k0 + 16;
        float m0 = pM[k0], h0 = pH[k0];
        aM[j] = f2bf((k0 < CIN) ? m0 : 0.f);
        aH[j] = f2bf((k0 < CIN) ? h0 : 0.f);
        float m1 = (k1 < CIN) ? pM[k1] : 0.f;
        float h1 = (k1 < CIN) ? pH[k1] : 0.f;
        aM[j + 8] = f2bf(m1);
        aH[j + 8] = f2bf(h1);
      }
      // B 32x16 bf16: lane holds column N = nB + (lane&15); element j has the
      // same K as A (kb + 8*lhi + j, then kb + 16 + 8*lhi + j).
      v16bf bL, bR;
#pragma unroll
      for (int j = 0; j < 8; ++j) {
        const int k0 = kb + lhi * 8 + j;     // always < KP
        const int k1 = k0 + 16;
        bL[j]     = wbl[k0 * NP + nB + lrow];
        bR[j]     = wbr[k0 * NP + nB + lrow];
        bL[j + 8] = wbl[k1 * NP + nB + lrow];
        bR[j + 8] = wbr[k1 * NP + nB + lrow];
      }
      acc = __builtin_amdgcn_wmma_f32_16x16x32_bf16(false, aM, false, bL,
                                                    (short)0, acc, false, false);
      acc = __builtin_amdgcn_wmma_f32_16x16x32_bf16(false, aH, false, bR,
                                                    (short)0, acc, false, false);
    }
    // D 16x16 f32: VGPR r -> M = mB + r + 8*lhi; N = nB + lrow
#pragma unroll
    for (int r = 0; r < 8; ++r) {
      const int m  = mB + r + lhi * 8;
      const int nn = nB + lrow;
      float v = ELU ? eluf(acc[r]) : acc[r];
      if (COUT == NP || nn < COUT)
        hout[m * COUT + nn] = v;
    }
  }
  __syncthreads();
}

// ---------------------------------------------------------------------------
// average pooling by cluster labels: (BT,NIN,C) -> (BT,NC,C)
// ---------------------------------------------------------------------------
template <int NIN, int NC, int C>
static __device__ void pool(const float* hin, float* hout, const int* labels,
                            float* cnt) {
  zero_lds(hout, BT * NC * C);
  zero_lds(cnt, NC);
  __syncthreads();
  for (int i = threadIdx.x; i < NIN; i += NTHR) atomicAdd(&cnt[labels[i]], 1.f);
  for (int t = threadIdx.x; t < BT * NIN; t += NTHR) {
    int node = t % NIN;
    int s    = t / NIN;
    int cl = labels[node];
    const float* src = &hin[t * C];
    float*       dst = &hout[(s * NC + cl) * C];
#pragma unroll
    for (int c = 0; c < C; ++c) atomicAdd(&dst[c], src[c]);
  }
  __syncthreads();
  for (int t = threadIdx.x; t < BT * NC; t += NTHR) {
    int cl = t % NC;
    float inv = 1.0f / fmaxf(cnt[cl], 1.0f);
#pragma unroll
    for (int c = 0; c < C; ++c) hout[t * C + c] *= inv;
  }
  __syncthreads();
}

// ---------------------------------------------------------------------------
// inverse-distance kNN interpolation: (BT,NX,C) -> (BT,NY,C)
// ---------------------------------------------------------------------------
template <int NX, int NY, int C>
static __device__ void interp(const float* hin, float* hout, const int* ai, int A,
                              const float* posx, const float* posy,
                              float* den, float* warr) {
  zero_lds(hout, BT * NY * C);
  zero_lds(den, NY);
  __syncthreads();
  for (int a = threadIdx.x; a < A; a += NTHR) {
    int yi = ai[a], xi = ai[A + a];
    float dx = posx[xi * 2]     - posy[yi * 2];
    float dy = posx[xi * 2 + 1] - posy[yi * 2 + 1];
    float w = 1.0f / fmaxf(dx * dx + dy * dy, 1e-16f);
    warr[a] = w;
    atomicAdd(&den[yi], w);
  }
  __syncthreads();
  for (int t = threadIdx.x; t < BT * A; t += NTHR) {
    int a = t % A;
    int s = t / A;
    int yi = ai[a], xi = ai[A + a];
    float w = warr[a];
    const float* src = &hin[(s * NX + xi) * C];
    float*       dst = &hout[(s * NY + yi) * C];
#pragma unroll
    for (int c = 0; c < C; ++c) atomicAdd(&dst[c], w * src[c]);
  }
  __syncthreads();
  for (int t = threadIdx.x; t < BT * NY; t += NTHR) {
    int yi = t % NY;
    float inv = 1.0f / den[yi];
#pragma unroll
    for (int c = 0; c < C; ++c) hout[t * C + c] *= inv;
  }
  __syncthreads();
}

#define SWAPAB() do { float* _t = A; A = Bb; Bb = _t; } while (0)

__global__ __launch_bounds__(NTHR)
void gunet_fused_kernel(Params p) {
  extern __shared__ float smem[];
  float* bufA = smem;
  float* bufB = bufA + BUFF;
  float* bufM = bufB + BUFF;
  float* aux  = bufM + BUFF;            // deg / cnt / den / z-scratch
  float* warr = aux + AUXF;             // kNN weights (<= 948)
  bf16*  wbl  = (bf16*)(warr + WARRF);  // staged Wl, bf16 B-layout (32B aligned)
  bf16*  wbr  = wbl + 4096;

  const int sBase = blockIdx.x * BT;
  float* A  = bufA;
  float* Bb = bufB;

  // ---- preprocess: h = (x - 1) / 7.5, layout (BT, 316, 1) ----
  for (int t = threadIdx.x; t < BT * 316; t += NTHR) {
    int s = t / 316, node = t % 316;
    A[t] = (p.x[(size_t)(sBase + s) * 316 + node] - 1.0f) * (1.0f / 7.5f);
  }
  __syncthreads();

  // ---- encoder level 1 (n=316): 1->8 (scalar), 8->8 (WMMA) ----
  aggregate<316, 1>(A, bufM, aux, p.e[0], p.E[0]);
  {
    const float* wl = p.wl[0]; const float* wr = p.wr[0];
    for (int t = threadIdx.x; t < BT * 316 * 8; t += NTHR) {
      int nn = t & 7; int m = t >> 3;
      Bb[t] = eluf(bufM[m] * wl[nn] + A[m] * wr[nn]);
    }
    __syncthreads();
  }
  SWAPAB();

  aggregate<316, 8>(A, bufM, aux, p.e[0], p.E[0]);
  sage_gemm<316, 8, 8, true>(bufM, A, p.wl[1], p.wr[1], Bb, wbl, wbr);  SWAPAB();

  pool<316, 158, 8>(A, Bb, p.lab[0], aux);                              SWAPAB();

  // ---- level 2 (n=158): 8->16, 16->16 ----
  aggregate<158, 8>(A, bufM, aux, p.e[1], p.E[1]);
  sage_gemm<158, 8, 16, true>(bufM, A, p.wl[2], p.wr[2], Bb, wbl, wbr); SWAPAB();
  aggregate<158, 16>(A, bufM, aux, p.e[1], p.E[1]);
  sage_gemm<158, 16, 16, true>(bufM, A, p.wl[3], p.wr[3], Bb, wbl, wbr); SWAPAB();

  pool<158, 40, 16>(A, Bb, p.lab[1], aux);                              SWAPAB();

  // ---- level 3 (n=40): 16->32, 32->32 ----
  aggregate<40, 16>(A, bufM, aux, p.e[2], p.E[2]);
  sage_gemm<40, 16, 32, true>(bufM, A, p.wl[4], p.wr[4], Bb, wbl, wbr); SWAPAB();
  aggregate<40, 32>(A, bufM, aux, p.e[2], p.E[2]);
  sage_gemm<40, 32, 32, true>(bufM, A, p.wl[5], p.wr[5], Bb, wbl, wbr); SWAPAB();

  pool<40, 10, 32>(A, Bb, p.lab[2], aux);                               SWAPAB();

  // ---- level 4 (n=10): 32->64, 64->64 ----
  aggregate<10, 32>(A, bufM, aux, p.e[3], p.E[3]);
  sage_gemm<10, 32, 64, true>(bufM, A, p.wl[6], p.wr[6], Bb, wbl, wbr); SWAPAB();
  aggregate<10, 64>(A, bufM, aux, p.e[3], p.E[3]);
  sage_gemm<10, 64, 64, true>(bufM, A, p.wl[7], p.wr[7], Bb, wbl, wbr); SWAPAB();

  pool<10, 2, 64>(A, Bb, p.lab[3], aux);                                SWAPAB();

  // ---- bottleneck MLP: (BT,128)@w5^T -> (BT,8); @w6^T -> elu -> (BT,128) ----
  {
    for (int t = threadIdx.x; t < BT * 8; t += NTHR) {
      int s = t >> 3, o = t & 7;
      float acc = 0.f;
      for (int k = 0; k < 128; ++k) acc += A[s * 128 + k] * p.w5[o * 128 + k];
      aux[t] = acc;                       // z (no ELU per reference)
    }
    __syncthreads();
    for (int t = threadIdx.x; t < BT * 128; t += NTHR) {
      int s = t >> 7, j = t & 127;
      float acc = 0.f;
#pragma unroll
      for (int o = 0; o < 8; ++o) acc += aux[s * 8 + o] * p.w6[j * 8 + o];
      Bb[t] = eluf(acc);
    }
    __syncthreads();
  }
  SWAPAB();

  // ---- decoder ----
  interp<2, 10, 64>(A, Bb, p.ai[0], p.A[0], p.pos[4], p.pos[3], aux, warr);   SWAPAB();
  aggregate<10, 64>(A, bufM, aux, p.e[3], p.E[3]);
  sage_gemm<10, 64, 64, true>(bufM, A, p.wl[8], p.wr[8], Bb, wbl, wbr);   SWAPAB();
  aggregate<10, 64>(A, bufM, aux, p.e[3], p.E[3]);
  sage_gemm<10, 64, 32, true>(bufM, A, p.wl[9], p.wr[9], Bb, wbl, wbr);   SWAPAB();

  interp<10, 40, 32>(A, Bb, p.ai[1], p.A[1], p.pos[3], p.pos[2], aux, warr);  SWAPAB();
  aggregate<40, 32>(A, bufM, aux, p.e[2], p.E[2]);
  sage_gemm<40, 32, 32, true>(bufM, A, p.wl[10], p.wr[10], Bb, wbl, wbr); SWAPAB();
  aggregate<40, 32>(A, bufM, aux, p.e[2], p.E[2]);
  sage_gemm<40, 32, 16, true>(bufM, A, p.wl[11], p.wr[11], Bb, wbl, wbr); SWAPAB();

  interp<40, 158, 16>(A, Bb, p.ai[2], p.A[2], p.pos[2], p.pos[1], aux, warr); SWAPAB();
  aggregate<158, 16>(A, bufM, aux, p.e[1], p.E[1]);
  sage_gemm<158, 16, 16, true>(bufM, A, p.wl[12], p.wr[12], Bb, wbl, wbr); SWAPAB();
  aggregate<158, 16>(A, bufM, aux, p.e[1], p.E[1]);
  sage_gemm<158, 16, 8, true>(bufM, A, p.wl[13], p.wr[13], Bb, wbl, wbr);  SWAPAB();

  interp<158, 316, 8>(A, Bb, p.ai[3], p.A[3], p.pos[1], p.pos[0], aux, warr); SWAPAB();
  aggregate<316, 8>(A, bufM, aux, p.e[0], p.E[0]);
  sage_gemm<316, 8, 8, true>(bufM, A, p.wl[14], p.wr[14], Bb, wbl, wbr);   SWAPAB();

  // ---- final SAGE 8->1 (scalar, no ELU) + scale + trimmed store ----
  aggregate<316, 8>(A, bufM, aux, p.e[0], p.E[0]);
  {
    const float* wl = p.wl[15]; const float* wr = p.wr[15];
    for (int m = threadIdx.x; m < BT * 316; m += NTHR) {
      float acc = 0.f;
#pragma unroll
      for (int k = 0; k < 8; ++k)
        acc += bufM[m * 8 + k] * wl[k] + A[m * 8 + k] * wr[k];
      float v = acc * 7.5f + 1.0f;
      int node = m % 316; int s = m / 316;
      if (node >= 30 && node < 286)
        p.out[(size_t)(sBase + s) * 256 + (node - 30)] = v;
    }
  }
}

extern "C" void kernel_launch(void* const* d_in, const int* in_sizes, int n_in,
                              void* d_out, int out_size, void* d_ws, size_t ws_size,
                              hipStream_t stream) {
  (void)n_in; (void)d_ws; (void)ws_size; (void)out_size;
  Params p;
  p.x = (const float*)d_in[0];
  for (int i = 0; i < 4; ++i) { p.e[i]  = (const int*)d_in[1 + i];  p.E[i] = in_sizes[1 + i] / 2; }
  for (int i = 0; i < 4; ++i)   p.lab[i] = (const int*)d_in[5 + i];
  for (int i = 0; i < 5; ++i)   p.pos[i] = (const float*)d_in[9 + i];
  for (int i = 0; i < 4; ++i) { p.ai[i] = (const int*)d_in[14 + i]; p.A[i] = in_sizes[14 + i] / 2; }
  for (int i = 0; i < 16; ++i) {
    p.wl[i] = (const float*)d_in[18 + 2 * i];
    p.wr[i] = (const float*)d_in[19 + 2 * i];
  }
  p.w5  = (const float*)d_in[50];
  p.w6  = (const float*)d_in[51];
  p.out = (float*)d_out;

  const int B = in_sizes[0] / 316;
  const size_t smemBytes = (size_t)SMEM_FLOATS * sizeof(float);
  hipFuncSetAttribute((const void*)gunet_fused_kernel,
                      hipFuncAttributeMaxDynamicSharedMemorySize, (int)smemBytes);
  hipLaunchKernelGGL(gunet_fused_kernel, dim3(B / BT), dim3(NTHR), smemBytes, stream, p);
}
